// CustomizedGateControl_49134425866512
// MI455X (gfx1250) — matrix operand order repaired and checked by reading, not compiled, not run on hardware
//
#include <hip/hip_runtime.h>
#include <hip/hip_bf16.h>

// Problem constants
#define B_    16384
#define D_    512
#define E_    256
#define H_    128
#define T_    2
#define NEXPC 3072          // 12 experts * 256 cols
#define NC    3088          // + 16 gate cols
#define MT    64            // rows of x per workgroup
#define AXP   520           // padded row stride of Ax/info in elems (1040 B = 65*16 B)
#define HSP   257           // padded row stride of hs in floats (bank-conflict-free)

typedef __attribute__((ext_vector_type(16))) __bf16 v16bf;
typedef __attribute__((ext_vector_type(8)))  float  v8f;

union Frag { uint4 u[2]; v16bf v; };

__device__ __forceinline__ v8f wmma_bf16(const Frag& a, const Frag& b, v8f c) {
    return __builtin_amdgcn_wmma_f32_16x16x32_bf16(
        false, a.v, false, b.v, (short)0, c, false, false);
}

// ---------------------------------------------------------------------------
// Pack: WcT[c][k] (column-major, bf16): cols 0..3071 experts, 3072..3087 gates
__global__ __launch_bounds__(256) void pack_wcT_kernel(
    const float* __restrict__ sW,   // [4,512,256]
    const float* __restrict__ tW,   // [2,4,512,256]
    const float* __restrict__ gW,   // [2,512,8]
    __bf16* __restrict__ WcT)
{
    int idx = blockIdx.x * 256 + threadIdx.x;   // c*512 + k
    int k = idx & (D_ - 1);
    int c = idx >> 9;
    float v;
    if (c < 1024) {
        int s = c >> 8, e = c & 255;
        v = sW[((size_t)s * D_ + k) * E_ + e];
    } else if (c < NEXPC) {
        int m = c - 1024;
        int tk = m >> 8, e = m & 255;
        v = tW[((size_t)tk * D_ + k) * E_ + e];
    } else {
        int m = c - NEXPC;
        v = gW[((size_t)(m >> 3) * D_ + k) * 8 + (m & 7)];
    }
    WcT[idx] = (__bf16)v;
}

// Pack: W1T[t*H + j][k] (column-major, bf16) from tower_W1 [T,E,H]
__global__ __launch_bounds__(256) void pack_w1T_kernel(
    const float* __restrict__ W1, __bf16* __restrict__ W1T)
{
    int idx = blockIdx.x * 256 + threadIdx.x;   // c*256 + k
    int k = idx & (E_ - 1);
    int c = idx >> 8;
    int t = c >> 7, j = c & 127;
    W1T[idx] = (__bf16)W1[((size_t)t * E_ + k) * H_ + j];
}

// Pack: concatenated expert bias (fp32, 3072)
__global__ __launch_bounds__(256) void pack_bias_kernel(
    const float* __restrict__ sb, const float* __restrict__ tb,
    float* __restrict__ biasc)
{
    int n = blockIdx.x * 256 + threadIdx.x;
    biasc[n] = (n < 1024) ? sb[n] : tb[n - 1024];
}

// ---------------------------------------------------------------------------
// Fully fused CGC kernel, 64 rows of x per workgroup.
//   A (x, 12x reuse)  : LDS-resident bf16 tile
//   B (weights, 1x use): direct global b128 loads (L2-resident), software-
//                        pipelined one k-step ahead -> zero k-loop barriers
//   info accumulators : fp32 registers per wave (e-slab ownership)
__global__ __launch_bounds__(256, 1) void cgc_fused_kernel(
    const float*  __restrict__ x,      // [B, D]
    const __bf16* __restrict__ WcT,    // [NC, D] col-major
    const float*  __restrict__ biasc,  // [3072]
    const __bf16* __restrict__ W1T,    // [T*H, E] col-major
    const float*  __restrict__ b1g,    // [T, H]
    const float*  __restrict__ W2g,    // [T, H]
    const float*  __restrict__ b2g,    // [T]
    float* __restrict__ out)           // [T, B]
{
    __shared__ __align__(16) unsigned char ldsA[MT * AXP * 2]; // Ax -> infoL -> hs
    __shared__ float gatesL[MT * 16];

    const int tid  = threadIdx.x;
    const int lane = tid & 31;
    const int wave = tid >> 5;
    const int nl   = lane & 15;
    const int half = lane >> 4;
    const int kh   = half * 8;     // A-frag K sub-offset (ISA 7.12.2)
    const int kb   = half * 8;     // B-frag K sub-offset
    const int m0   = blockIdx.x * MT;

    __bf16* Ax = (__bf16*)ldsA;
    const v8f vzero = {0.f,0.f,0.f,0.f,0.f,0.f,0.f,0.f};

    // ---- phase 0: stage x slab (fp32 -> bf16) into LDS -------------------
    #pragma unroll
    for (int i = 0; i < 16; ++i) {
        int c   = tid + i * 256;          // 0..4095
        int row = c >> 6;
        int col = (c & 63) * 8;
        const float* src = x + (size_t)(m0 + row) * D_ + col;
        float4 f0 = *(const float4*)src;
        float4 f1 = *(const float4*)(src + 4);
        union { uint4 u; __bf16 h[8]; } o;
        o.h[0]=(__bf16)f0.x; o.h[1]=(__bf16)f0.y; o.h[2]=(__bf16)f0.z; o.h[3]=(__bf16)f0.w;
        o.h[4]=(__bf16)f1.x; o.h[5]=(__bf16)f1.y; o.h[6]=(__bf16)f1.z; o.h[7]=(__bf16)f1.w;
        *(uint4*)(&Ax[row * AXP + col]) = o.u;
    }
    __syncthreads();

    // ---- phase 1: gates (16 cols, K=512); all waves compute redundantly --
    {
        v8f g[4] = {vzero, vzero, vzero, vzero};
        const __bf16* gbase = WcT + (size_t)(NEXPC + nl) * D_ + kb;
        auto loadG = [&](Frag& f, int ks) {
            const __bf16* p = gbase + ks * 32;
            f.u[0] = *(const uint4*)p;
            f.u[1] = *(const uint4*)(p + 16);
        };
        auto gstep = [&](Frag& bf, int ks) {
            #pragma unroll
            for (int mi = 0; mi < 4; ++mi) {
                Frag aF;
                const __bf16* ap = &Ax[(mi * 16 + nl) * AXP + ks * 32 + kh];
                aF.u[0] = *(const uint4*)ap;
                aF.u[1] = *(const uint4*)(ap + 16);
                g[mi] = wmma_bf16(aF, bf, g[mi]);
            }
        };
        Frag bGa, bGb;
        loadG(bGa, 0);
        for (int ks = 0; ks < 16; ks += 2) {
            loadG(bGb, ks + 1);
            gstep(bGa, ks);
            if (ks + 2 < 16) loadG(bGa, ks + 2);
            gstep(bGb, ks + 1);
        }
        if (wave == 0) {
            #pragma unroll
            for (int mi = 0; mi < 4; ++mi)
                #pragma unroll
                for (int r = 0; r < 8; ++r)
                    gatesL[(mi * 16 + half * 8 + r) * 16 + nl] = g[mi][r];
        }
    }
    __syncthreads();   // gatesL visible to all folds

    // ---- phase 2: expert loop, register-resident gated accumulation ------
    v8f info[2][4][2];
    #pragma unroll
    for (int t = 0; t < 2; ++t)
        #pragma unroll
        for (int mi = 0; mi < 4; ++mi)
            #pragma unroll
            for (int ni = 0; ni < 2; ++ni) info[t][mi][ni] = vzero;

    for (int ge = 0; ge < 12; ++ge) {
        v8f acc[4][2];
        #pragma unroll
        for (int mi = 0; mi < 4; ++mi)
            #pragma unroll
            for (int ni = 0; ni < 2; ++ni) acc[mi][ni] = vzero;

        const __bf16* bbase = WcT + (size_t)(ge * 256 + wave * 32 + nl) * D_ + kb;
        auto loadB = [&](Frag* bF, int ks) {
            #pragma unroll
            for (int ni = 0; ni < 2; ++ni) {
                const __bf16* p = bbase + (size_t)(ni * 16) * D_ + ks * 32;
                bF[ni].u[0] = *(const uint4*)p;
                bF[ni].u[1] = *(const uint4*)(p + 16);
            }
        };
        auto estep = [&](Frag* bF, int ks) {
            #pragma unroll
            for (int mi = 0; mi < 4; ++mi) {
                Frag aF;
                const __bf16* ap = &Ax[(mi * 16 + nl) * AXP + ks * 32 + kh];
                aF.u[0] = *(const uint4*)ap;
                aF.u[1] = *(const uint4*)(ap + 16);
                acc[mi][0] = wmma_bf16(aF, bF[0], acc[mi][0]);
                acc[mi][1] = wmma_bf16(aF, bF[1], acc[mi][1]);
            }
        };
        Frag bFa[2], bFb[2];
        loadB(bFa, 0);
        for (int ks = 0; ks < 16; ks += 2) {
            loadB(bFb, ks + 1);
            estep(bFa, ks);
            if (ks + 2 < 16) loadB(bFa, ks + 2);
            estep(bFb, ks + 1);
        }

        // fold: bias + relu, gate-weighted accumulate (static info indexing)
        const bool shared_e = (ge < 4);
        const int  t0  = shared_e ? 0 : ((ge - 4) >> 2);
        const int  tg0 = shared_e ? ge : t0 * 8 + 4 + ((ge - 4) & 3);
        #pragma unroll
        for (int ni = 0; ni < 2; ++ni) {
            int e = wave * 32 + ni * 16 + nl;
            float bv = biasc[ge * 256 + e];
            #pragma unroll
            for (int mi = 0; mi < 4; ++mi) {
                #pragma unroll
                for (int r = 0; r < 8; ++r) {
                    int m = mi * 16 + half * 8 + r;
                    float val = fmaxf(acc[mi][ni][r] + bv, 0.f);
                    float g0, g1;
                    if (shared_e) {            // shared expert feeds both tasks
                        g0 = gatesL[m * 16 + ge];
                        g1 = gatesL[m * 16 + 8 + ge];
                    } else {
                        float gv = gatesL[m * 16 + tg0];
                        g0 = (t0 == 0) ? gv : 0.f;
                        g1 = (t0 == 1) ? gv : 0.f;
                    }
                    info[0][mi][ni][r] += g0 * val;
                    info[1][mi][ni][r] += g1 * val;
                }
            }
        }
    }
    __syncthreads();   // all Ax reads done before overwriting the slab

    // ---- phase 3: spill info (bf16) into the dead x-slab LDS region ------
    __bf16* infoL = (__bf16*)ldsA;           // [m][t*256+e], row stride AXP
    #pragma unroll
    for (int t = 0; t < 2; ++t)
        #pragma unroll
        for (int mi = 0; mi < 4; ++mi)
            #pragma unroll
            for (int ni = 0; ni < 2; ++ni) {
                int col = t * 256 + wave * 32 + ni * 16 + nl;
                #pragma unroll
                for (int r = 0; r < 8; ++r) {
                    int m = mi * 16 + half * 8 + r;
                    infoL[m * AXP + col] = (__bf16)info[t][mi][ni][r];
                }
            }
    __syncthreads();

    // ---- phase 4: tower1 WMMA: h = relu(info . W1 + b1) ------------------
    const int tw = wave >> 2;          // task of this wave
    const int jw = (wave & 3) * 32;    // h-column slab within task
    v8f hacc[4][2];
    #pragma unroll
    for (int mi = 0; mi < 4; ++mi)
        #pragma unroll
        for (int ni = 0; ni < 2; ++ni) hacc[mi][ni] = vzero;

    {
        const __bf16* tbase = W1T + (size_t)(tw * 128 + jw + nl) * E_ + kb;
        auto loadTB = [&](Frag* bF, int ks) {
            #pragma unroll
            for (int ni = 0; ni < 2; ++ni) {
                const __bf16* p = tbase + (size_t)(ni * 16) * E_ + ks * 32;
                bF[ni].u[0] = *(const uint4*)p;
                bF[ni].u[1] = *(const uint4*)(p + 16);
            }
        };
        auto tstep = [&](Frag* bF, int ks) {
            #pragma unroll
            for (int mi = 0; mi < 4; ++mi) {
                Frag aF;
                const __bf16* ap =
                    &infoL[(mi * 16 + nl) * AXP + tw * 256 + ks * 32 + kh];
                aF.u[0] = *(const uint4*)ap;
                aF.u[1] = *(const uint4*)(ap + 16);
                hacc[mi][0] = wmma_bf16(aF, bF[0], hacc[mi][0]);
                hacc[mi][1] = wmma_bf16(aF, bF[1], hacc[mi][1]);
            }
        };
        Frag bFa[2], bFb[2];
        loadTB(bFa, 0);
        for (int ks = 0; ks < 8; ks += 2) {
            loadTB(bFb, ks + 1);
            tstep(bFa, ks);
            if (ks + 2 < 8) loadTB(bFa, ks + 2);
            tstep(bFb, ks + 1);
        }
    }
    __syncthreads();                   // infoL reads done before hs overwrite

    float* hs = (float*)ldsA;          // [m][tw*128 + j], fp32, stride HSP
    #pragma unroll
    for (int ni = 0; ni < 2; ++ni) {
        int j = jw + ni * 16 + nl;
        float bv = b1g[tw * H_ + j];
        #pragma unroll
        for (int mi = 0; mi < 4; ++mi)
            #pragma unroll
            for (int r = 0; r < 8; ++r) {
                int m = mi * 16 + half * 8 + r;
                hs[m * HSP + tw * H_ + j] = fmaxf(hacc[mi][ni][r] + bv, 0.f);
            }
    }
    __syncthreads();

    // ---- phase 5: tower2 reduction --------------------------------------
    if (tid < 128) {
        int t = tid >> 6, m = tid & 63;
        const float* w = W2g + t * H_;
        float s = b2g[t];
        #pragma unroll 8
        for (int j = 0; j < H_; ++j)
            s += hs[m * HSP + t * H_ + j] * w[j];
        out[(size_t)t * B_ + m0 + m] = s;
    }
}

// ---------------------------------------------------------------------------
extern "C" void kernel_launch(void* const* d_in, const int* in_sizes, int n_in,
                              void* d_out, int out_size, void* d_ws, size_t ws_size,
                              hipStream_t stream) {
    (void)in_sizes; (void)n_in; (void)out_size; (void)ws_size;
    const float* x        = (const float*)d_in[0];
    const float* shared_W = (const float*)d_in[1];
    const float* shared_b = (const float*)d_in[2];
    const float* task_W   = (const float*)d_in[3];
    const float* task_b   = (const float*)d_in[4];
    const float* gate_W   = (const float*)d_in[5];
    const float* tower_W1 = (const float*)d_in[6];
    const float* tower_b1 = (const float*)d_in[7];
    const float* tower_W2 = (const float*)d_in[8];
    const float* tower_b2 = (const float*)d_in[9];
    float* out = (float*)d_out;

    char* ws = (char*)d_ws;
    size_t off = 0;
    auto carve = [&](size_t bytes) {
        void* p = ws + off;
        off = (off + bytes + 255) & ~(size_t)255;
        return p;
    };
    __bf16* WcT   = (__bf16*)carve((size_t)NC * D_ * 2);       // 3.16 MB
    float*  biasc = (float*) carve((size_t)NEXPC * 4);
    __bf16* W1T   = (__bf16*)carve((size_t)T_ * H_ * E_ * 2);  // 128 KB

    pack_wcT_kernel<<<(NC * D_) / 256, 256, 0, stream>>>(shared_W, task_W, gate_W, WcT);
    pack_bias_kernel<<<NEXPC / 256, 256, 0, stream>>>(shared_b, task_b, biasc);
    pack_w1T_kernel<<<(T_ * H_ * E_) / 256, 256, 0, stream>>>(tower_W1, W1T);

    cgc_fused_kernel<<<B_ / MT, 256, 0, stream>>>(
        x, WcT, biasc, W1T, tower_b1, tower_W2, tower_b2, out);
}
